// SegSmall_5119601017102
// MI455X (gfx1250) — matrix operand order repaired and checked by reading, not compiled, not run on hardware
//
#include <hip/hip_runtime.h>

typedef _Float16 half_t;
typedef __attribute__((ext_vector_type(16))) _Float16 v16h;
typedef __attribute__((ext_vector_type(8)))  _Float16 v8h;
typedef __attribute__((ext_vector_type(8)))  float    v8f;

#define BB 2
#define NN 10000
#define BN_EPS 1e-5f

// ===================== device kernels =====================

// points [B,3,N] -> input_pts [B,N,3]
__global__ void k_transpose_pts(const float* __restrict__ src,
                                float* __restrict__ dst) {
  int t = blockIdx.x * blockDim.x + threadIdx.x;
  if (t >= BB * NN) return;
  int b = t / NN, i = t - b * NN;
  const float* s = src + (size_t)b * 3 * NN;
  float* d = dst + (size_t)t * 3;
  d[0] = s[i];
  d[1] = s[NN + i];
  d[2] = s[2 * NN + i];
}

__global__ void k_fill_f16(half_t* __restrict__ p, int n) {
  int t = blockIdx.x * blockDim.x + threadIdx.x;
  if (t < n) p[t] = (half_t)0.0f;
}

// brute-force K nearest neighbors; queries & points are prefixes of the same
// per-batch point array P [B, NN, 3]
__global__ void k_knn(const float* __restrict__ P, int Np, int M, int K,
                      int* __restrict__ idx) {
  int t = blockIdx.x * blockDim.x + threadIdx.x;
  if (t >= BB * M) return;
  int b = t / M, m = t - b * M;
  const float* base = P + (size_t)b * NN * 3;
  float qx = base[3 * m + 0], qy = base[3 * m + 1], qz = base[3 * m + 2];
  float bd[16];
  int bi[16];
  for (int j = 0; j < 16; ++j) { bd[j] = 3.0e38f; bi[j] = 0; }
  for (int n = 0; n < Np; ++n) {
    __builtin_prefetch(base + 3 * (n + 32), 0, 0);
    float dx = base[3 * n + 0] - qx;
    float dy = base[3 * n + 1] - qy;
    float dz = base[3 * n + 2] - qz;
    float d2 = dx * dx + dy * dy + dz * dz;
    if (d2 < bd[K - 1]) {
      int j = K - 1;
      while (j > 0 && bd[j - 1] > d2) { bd[j] = bd[j - 1]; bi[j] = bi[j - 1]; --j; }
      bd[j] = d2; bi[j] = n;
    }
  }
  int* o = idx + (size_t)t * K;
  for (int k = 0; k < K; ++k) o[k] = bi[k];
}

// Build MLP input D rows: for each (b,m,k): 48 values (nbr/maxi - centers),
// stored f16 at ld=64 with cols 48..63 zeroed.
__global__ void k_build_d(const float* __restrict__ P, const int* __restrict__ idx,
                          const float* __restrict__ centers, int M, int K,
                          half_t* __restrict__ D) {
  int t = blockIdx.x * blockDim.x + threadIdx.x;
  if (t >= BB * M) return;
  int b = t / M, m = t - b * M;
  const float* base = P + (size_t)b * NN * 3;
  float qx = base[3 * m + 0], qy = base[3 * m + 1], qz = base[3 * m + 2];
  const int* id = idx + (size_t)t * K;
  float ox[16], oy[16], oz[16];
  float maxd2 = 0.0f;
  for (int k = 0; k < K; ++k) {
    int n = id[k];
    float dx = base[3 * n + 0] - qx;
    float dy = base[3 * n + 1] - qy;
    float dz = base[3 * n + 2] - qz;
    ox[k] = dx; oy[k] = dy; oz[k] = dz;
    float d2 = dx * dx + dy * dy + dz * dz;
    maxd2 = fmaxf(maxd2, d2);
  }
  float maxi = sqrtf(maxd2);
  if (maxi == 0.0f) maxi = 1.0f;
  float inv = 1.0f / maxi;
  for (int k = 0; k < K; ++k) {
    half_t* row = D + ((size_t)t * K + k) * 64;
    float nb0 = ox[k] * inv, nb1 = oy[k] * inv, nb2 = oz[k] * inv;
    for (int c = 0; c < 16; ++c) {
      row[c]      = (half_t)(nb0 - centers[c]);
      row[16 + c] = (half_t)(nb1 - centers[16 + c]);
      row[32 + c] = (half_t)(nb2 - centers[32 + c]);
    }
    for (int c = 48; c < 64; ++c) row[c] = (half_t)0.0f;
  }
}

// Pack a f32 weight matrix into WMMA B-fragment order.
// Logical B is [Kr x Nr]; if transB, element (k,n) = src[n*Kr + k] else src[k*Nr + n].
// dst layout: tile (nt*Ktiles + kt) * 512 + lane*16 + i, where
//   k = kt*32 + (i<8?0:16) + (lane<16?0:8) + (i&7) ; n = nt*16 + (lane&15)
__global__ void k_pack_b(const float* __restrict__ src, int Kr, int Nr, int transB,
                         int Ktiles, int Ntiles, half_t* __restrict__ dst) {
  int t = blockIdx.x * blockDim.x + threadIdx.x;
  int total = Ktiles * Ntiles * 512;
  if (t >= total) return;
  int i = t & 15;
  int lane = (t >> 4) & 31;
  int tile = t >> 9;
  int kt = tile % Ktiles;
  int nt = tile / Ktiles;
  int kk = kt * 32 + ((i < 8) ? 0 : 16) + ((lane < 16) ? 0 : 8) + (i & 7);
  int n = nt * 16 + (lane & 15);
  float v = 0.0f;
  if (kk < Kr && n < Nr)
    v = transB ? src[(size_t)n * Kr + kk] : src[(size_t)kk * Nr + n];
  dst[t] = (half_t)v;
}

// One wave computes a 32x64 output block (2 M-tiles x 4 N-tiles, 8 accumulators)
// to raise WMMA arithmetic intensity (~21 FLOP/byte of fragment traffic).
// A: f16 row-major, rows padded to 32*gridDim.x, lda multiple of 32.
// Bp: packed fragments (tile (nt*Ktiles+kt)*512).  Ragged N handled by clamping
// B-tile pointers (redundant math, stores guarded).  Epilogue: *scale + bias,
// optional relu; writes f32 (bounds-guarded) and/or f16 (padded buffer).
__global__ void k_gemm_wmma(const half_t* __restrict__ A, int lda,
                            const half_t* __restrict__ Bp, int Ktiles, int Ntiles,
                            float* __restrict__ Cf, int ldc, int Mrows, int Ncols,
                            half_t* __restrict__ C16, int ldc16,
                            const float* __restrict__ bias,
                            float scale, int relu) {
  int tm = blockIdx.x;
  int tn4 = blockIdx.y * 4;
  int lane = threadIdx.x;          // 0..31, one full wave, uniform control flow
  int ncnt = Ntiles - tn4;
  if (ncnt > 4) ncnt = 4;
  int hi = (lane >> 4) & 1;
  int lc = lane & 15;
  const half_t* arow0 = A + (size_t)(tm * 32 + lc) * lda + hi * 8;
  const half_t* arow1 = arow0 + (size_t)16 * lda;
  int n1 = tn4 + 1, n2 = tn4 + 2, n3 = tn4 + 3;
  if (n1 >= Ntiles) n1 = Ntiles - 1;
  if (n2 >= Ntiles) n2 = Ntiles - 1;
  if (n3 >= Ntiles) n3 = Ntiles - 1;
  const half_t* bt0 = Bp + (size_t)tn4 * Ktiles * 512 + lane * 16;
  const half_t* bt1 = Bp + (size_t)n1 * Ktiles * 512 + lane * 16;
  const half_t* bt2 = Bp + (size_t)n2 * Ktiles * 512 + lane * 16;
  const half_t* bt3 = Bp + (size_t)n3 * Ktiles * 512 + lane * 16;
  v8f c00 = {0.f, 0.f, 0.f, 0.f, 0.f, 0.f, 0.f, 0.f};
  v8f c01 = c00, c02 = c00, c03 = c00;
  v8f c10 = c00, c11 = c00, c12 = c00, c13 = c00;
  for (int kt = 0; kt < Ktiles; ++kt) {
    v8h x0 = *(const v8h*)(arow0);
    v8h x1 = *(const v8h*)(arow0 + 16);
    v16h a0 = __builtin_shufflevector(x0, x1, 0, 1, 2, 3, 4, 5, 6, 7,
                                      8, 9, 10, 11, 12, 13, 14, 15);
    v8h y0 = *(const v8h*)(arow1);
    v8h y1 = *(const v8h*)(arow1 + 16);
    v16h a1 = __builtin_shufflevector(y0, y1, 0, 1, 2, 3, 4, 5, 6, 7,
                                      8, 9, 10, 11, 12, 13, 14, 15);
    v16h b0 = *(const v16h*)(bt0);
    v16h b1 = *(const v16h*)(bt1);
    v16h b2 = *(const v16h*)(bt2);
    v16h b3 = *(const v16h*)(bt3);
    c00 = __builtin_amdgcn_wmma_f32_16x16x32_f16(false, a0, false, b0, (short)0, c00, false, false);
    c10 = __builtin_amdgcn_wmma_f32_16x16x32_f16(false, a1, false, b0, (short)0, c10, false, false);
    c01 = __builtin_amdgcn_wmma_f32_16x16x32_f16(false, a0, false, b1, (short)0, c01, false, false);
    c11 = __builtin_amdgcn_wmma_f32_16x16x32_f16(false, a1, false, b1, (short)0, c11, false, false);
    c02 = __builtin_amdgcn_wmma_f32_16x16x32_f16(false, a0, false, b2, (short)0, c02, false, false);
    c12 = __builtin_amdgcn_wmma_f32_16x16x32_f16(false, a1, false, b2, (short)0, c12, false, false);
    c03 = __builtin_amdgcn_wmma_f32_16x16x32_f16(false, a0, false, b3, (short)0, c03, false, false);
    c13 = __builtin_amdgcn_wmma_f32_16x16x32_f16(false, a1, false, b3, (short)0, c13, false, false);
    arow0 += 32;
    arow1 += 32;
    bt0 += 512; bt1 += 512; bt2 += 512; bt3 += 512;
  }
  int rb = tm * 32 + hi * 8;
  auto store_tile = [&](const v8f& cc, int mt, int j) {
    int col = (tn4 + j) * 16 + lc;
    float bsv = 0.0f;
    if (bias != nullptr && col < Ncols) bsv = bias[col];
#pragma unroll
    for (int v = 0; v < 8; ++v) {
      float val = cc[v] * scale + bsv;
      if (relu) val = fmaxf(val, 0.0f);
      int r = rb + mt * 16 + v;
      if (C16) C16[(size_t)r * ldc16 + col] = (half_t)val;
      if (Cf && r < Mrows && col < Ncols) Cf[(size_t)r * ldc + col] = val;
    }
  };
  store_tile(c00, 0, 0);
  store_tile(c10, 1, 0);
  if (ncnt > 1) { store_tile(c01, 0, 1); store_tile(c11, 1, 1); }
  if (ncnt > 2) { store_tile(c02, 0, 2); store_tile(c12, 1, 2); }
  if (ncnt > 3) { store_tile(c03, 0, 3); store_tile(c13, 1, 3); }
}

// f[bm, c*16+n] = sum_k x[b, idx[bm,k], c] * H[bm*K+k, n]; writes f16 A-buffer.
// x == nullptr means features are all ones (first conv, Cin==1).
__global__ void k_einsum(const float* __restrict__ x, int Np, int Cin,
                         const int* __restrict__ idx, int M, int K,
                         const float* __restrict__ H,
                         half_t* __restrict__ F, int ldf) {
  int t = blockIdx.x * blockDim.x + threadIdx.x;
  if (t >= BB * M * Cin) return;
  int c = t % Cin;
  int bm = t / Cin;
  int b = bm / M;
  float acc[16];
#pragma unroll
  for (int n = 0; n < 16; ++n) acc[n] = 0.0f;
  const int* id = idx + (size_t)bm * K;
  for (int k = 0; k < K; ++k) {
    float f = (x != nullptr) ? x[((size_t)b * Np + id[k]) * Cin + c] : 1.0f;
    const float* h = H + ((size_t)bm * K + k) * 16;
#pragma unroll
    for (int n = 0; n < 16; ++n) acc[n] += f * h[n];
  }
  half_t* o = F + (size_t)bm * ldf + c * 16;
#pragma unroll
  for (int n = 0; n < 16; ++n) o[n] = (half_t)acc[n];
}

// batch-stats BN: one block per channel
__global__ void k_bn_stats(const float* __restrict__ X, int R, int C,
                           float* __restrict__ mean, float* __restrict__ rstd) {
  int c = blockIdx.x;
  __shared__ float s1[256];
  __shared__ float s2[256];
  float a = 0.0f, bsum = 0.0f;
  for (int r = threadIdx.x; r < R; r += blockDim.x) {
    float v = X[(size_t)r * C + c];
    a += v;
    bsum += v * v;
  }
  s1[threadIdx.x] = a;
  s2[threadIdx.x] = bsum;
  __syncthreads();
  for (int s = 128; s > 0; s >>= 1) {
    if ((int)threadIdx.x < s) {
      s1[threadIdx.x] += s1[threadIdx.x + s];
      s2[threadIdx.x] += s2[threadIdx.x + s];
    }
    __syncthreads();
  }
  if (threadIdx.x == 0) {
    float m = s1[0] / (float)R;
    float var = s2[0] / (float)R - m * m;
    mean[c] = m;
    rstd[c] = rsqrtf(var + BN_EPS);
  }
}

__global__ void k_bn_apply(const float* __restrict__ X, int R, int C,
                           const float* __restrict__ mean,
                           const float* __restrict__ rstd,
                           const float* __restrict__ gamma,
                           const float* __restrict__ beta,
                           float* __restrict__ Y, int ldy, int coff) {
  int t = blockIdx.x * blockDim.x + threadIdx.x;
  if (t >= R * C) return;
  int r = t / C, c = t - r * C;
  float v = (X[t] - mean[c]) * rstd[c] * gamma[c] + beta[c];
  Y[(size_t)r * ldy + coff + c] = fmaxf(v, 0.0f);
}

__global__ void k_copy_cols(const float* __restrict__ X, int R, int C,
                            float* __restrict__ Y, int ldy, int coff) {
  int t = blockIdx.x * blockDim.x + threadIdx.x;
  if (t >= R * C) return;
  int r = t / C, c = t - r * C;
  Y[(size_t)r * ldy + coff + c] = X[t];
}

// f32 [R x C] -> f16 [Rp x ldy] with zero padding
__global__ void k_f32_to_f16(const float* __restrict__ X, int R, int C,
                             half_t* __restrict__ Y, int Rp, int ldy) {
  int t = blockIdx.x * blockDim.x + threadIdx.x;
  if (t >= Rp * ldy) return;
  int r = t / ldy, c = t - r * ldy;
  float v = (r < R && c < C) ? X[(size_t)r * C + c] : 0.0f;
  Y[t] = (half_t)v;
}

// [B*N x 50] -> [B, 50, N]
__global__ void k_seg_transpose(const float* __restrict__ S, float* __restrict__ out) {
  int t = blockIdx.x * blockDim.x + threadIdx.x;
  if (t >= BB * 50 * NN) return;
  int n = t % NN;
  int s = (t / NN) % 50;
  int b = t / (50 * NN);
  out[t] = S[((size_t)b * NN + n) * 50 + s];
}

// ===================== host orchestration =====================

struct PCp { const float *weight, *centers, *l1w, *l1b, *l2w, *l2b, *l3w, *l3b; };
struct BNp { const float *gamma, *beta; };

struct Scratch {
  float* Pts;
  int* idx;
  half_t* D;
  half_t* H1;
  half_t* H2;
  float* H;
  half_t* F;
  float* RAW;
  half_t* Bp1;
  half_t* Bp2;
  half_t* Bp3;
  half_t* BpW;
  float* mean;
  float* rstd;
};

static inline int cdiv(int a, int b) { return (a + b - 1) / b; }
static inline int pad32(int a) { return cdiv(a, 32) * 32; }

static void run_ptconv(hipStream_t s, Scratch& w, int Np, int M, int Kn,
                       const float* x, int Cin, const PCp& p, int Cout,
                       float* rawOut) {
  int BM = BB * M;
  int R = BM * Kn;
  int Rp = pad32(R);
  k_knn<<<cdiv(BM, 128), 128, 0, s>>>(w.Pts, Np, M, Kn, w.idx);
  if (Rp > R)
    k_fill_f16<<<cdiv((Rp - R) * 64, 256), 256, 0, s>>>(w.D + (size_t)R * 64,
                                                        (Rp - R) * 64);
  k_build_d<<<cdiv(BM, 128), 128, 0, s>>>(w.Pts, w.idx, p.centers, M, Kn, w.D);
  // MLP layer 1: [Rp x 64(48 real)] x [64 x 32], relu, -> f16 H1
  k_pack_b<<<cdiv(2 * 2 * 512, 256), 256, 0, s>>>(p.l1w, 48, 32, 1, 2, 2, w.Bp1);
  k_gemm_wmma<<<dim3(Rp / 32, 1), 32, 0, s>>>(w.D, 64, w.Bp1, 2, 2,
      (float*)nullptr, 0, Rp, 32, w.H1, 32, p.l1b, 1.0f, 1);
  // MLP layer 2: [Rp x 32] x [32 x 16], relu, -> f16 H2 (ld 32, pad cols zeroed)
  k_fill_f16<<<cdiv(Rp * 32, 256), 256, 0, s>>>(w.H2, Rp * 32);
  k_pack_b<<<1, 256, 0, s>>>(p.l2w, 32, 16, 1, 1, 1, w.Bp2);
  k_gemm_wmma<<<dim3(Rp / 32, 1), 32, 0, s>>>(w.H1, 32, w.Bp2, 1, 1,
      (float*)nullptr, 0, Rp, 16, w.H2, 32, p.l2b, 1.0f, 1);
  // MLP layer 3: [Rp x 32(16 real)] x [32 x 16], relu, -> f32 H [R x 16]
  k_pack_b<<<1, 256, 0, s>>>(p.l3w, 16, 16, 1, 1, 1, w.Bp3);
  k_gemm_wmma<<<dim3(Rp / 32, 1), 32, 0, s>>>(w.H2, 32, w.Bp3, 1, 1,
      w.H, 16, R, 16, (half_t*)nullptr, 0, p.l3b, 1.0f, 1);
  // einsum: feats (gathered) x H -> F f16 [pad32(BM) x Fld]
  int Fld = Cin * 16;
  if (Fld & 31) Fld = (Fld + 31) & ~31;
  int BMp = pad32(BM);
  if ((Cin * 16) & 31)
    k_fill_f16<<<cdiv(BMp * Fld, 256), 256, 0, s>>>(w.F, BMp * Fld);
  k_einsum<<<cdiv(BM * Cin, 128), 128, 0, s>>>(x, Np, Cin, w.idx, M, Kn, w.H,
                                               w.F, Fld);
  // projection: [BMp x Fld] x weight[Cin*16 x Cout], scale 1/K -> f32 raw out
  int Kt = Fld / 32;
  int Nt = cdiv(Cout, 16);
  k_pack_b<<<cdiv(Kt * Nt * 512, 256), 256, 0, s>>>(p.weight, Cin * 16, Cout, 0,
                                                    Kt, Nt, w.BpW);
  k_gemm_wmma<<<dim3(BMp / 32, cdiv(Nt, 4)), 32, 0, s>>>(w.F, Fld, w.BpW, Kt, Nt,
      rawOut, Cout, BM, Cout, (half_t*)nullptr, 0, (const float*)nullptr,
      1.0f / (float)Kn, 0);
}

static void run_bn(hipStream_t s, Scratch& w, const float* Xraw, int R, int C,
                   const BNp& bnp, float* Y, int ldy, int coff) {
  k_bn_stats<<<C, 256, 0, s>>>(Xraw, R, C, w.mean, w.rstd);
  k_bn_apply<<<cdiv(R * C, 256), 256, 0, s>>>(Xraw, R, C, w.mean, w.rstd,
                                              bnp.gamma, bnp.beta, Y, ldy, coff);
}

extern "C" void kernel_launch(void* const* d_in, const int* in_sizes, int n_in,
                              void* d_out, int out_size, void* d_ws, size_t ws_size,
                              hipStream_t stream) {
  (void)out_size; (void)ws_size;
  // conv order: cv2,cv3,cv4,cv5,cv6,cv5d,cv4d,cv3d,cv2d,cv1d,conv_cls
  static const int cvCin[11]  = {1, 48, 48, 96, 96, 128, 352, 192, 96, 304, 128};
  static const int cvCout[11] = {48, 48, 96, 96, 128, 256, 96, 48, 256, 128, 256};
  // bn order: bn2,bn3,bn4,bn5,bn6,bn5d,bn4d,bn3d,bn2d,bn1d,bn6_conv
  static const int bnCh[11]   = {48, 48, 96, 96, 128, 256, 96, 48, 256, 128, 256};
  static const int bnOrd[11]  = {9, 0, 8, 1, 7, 2, 6, 3, 5, 4, 10};   // sorted name order
  static const int cvOrd[11]  = {10, 9, 0, 8, 1, 7, 2, 6, 3, 5, 4};   // sorted name order

  // Expected flat-size sequences for the two plausible flattening orders.
  int expIns[115];
  int k = 0;
  expIns[k++] = 60000;
  for (int c = 0; c < 11; ++c) {
    expIns[k++] = cvCin[c] * 16 * cvCout[c];
    expIns[k++] = 48; expIns[k++] = 1536; expIns[k++] = 32;
    expIns[k++] = 512; expIns[k++] = 16; expIns[k++] = 256; expIns[k++] = 16;
  }
  for (int b = 0; b < 11; ++b) { expIns[k++] = bnCh[b]; expIns[k++] = bnCh[b]; }
  expIns[k++] = 6400; expIns[k++] = 50; expIns[k++] = 4096; expIns[k++] = 16;

  int expSrt[115];
  k = 0;
  for (int j = 0; j < 11; ++j) { int b = bnOrd[j]; expSrt[k++] = bnCh[b]; expSrt[k++] = bnCh[b]; }
  expSrt[k++] = 16; expSrt[k++] = 4096;
  for (int j = 0; j < 11; ++j) {
    int c = cvOrd[j];
    expSrt[k++] = 48; expSrt[k++] = 32; expSrt[k++] = 1536; expSrt[k++] = 16;
    expSrt[k++] = 512; expSrt[k++] = 16; expSrt[k++] = 256;
    expSrt[k++] = cvCin[c] * 16 * cvCout[c];
  }
  expSrt[k++] = 50; expSrt[k++] = 6400; expSrt[k++] = 60000;

  bool useIns = true;
  if (n_in == 115) {
    bool mi = true, ms = true;
    for (int i = 0; i < 115; ++i) {
      if (in_sizes[i] != expIns[i]) mi = false;
      if (in_sizes[i] != expSrt[i]) ms = false;
    }
    if (!mi && ms) useIns = false;
  } else if (n_in > 0 && in_sizes[0] != 60000) {
    useIns = false;
  }

  const float* points = nullptr;
  PCp pc[11];
  BNp bn[11];
  const float *segw = nullptr, *segb = nullptr, *clsw = nullptr, *clsb = nullptr;
  int ii = 0;
  auto nextp = [&]() -> const float* { return (const float*)d_in[ii++]; };
  if (useIns) {
    points = nextp();
    for (int c = 0; c < 11; ++c) {
      PCp& p = pc[c];
      p.weight = nextp(); p.centers = nextp();
      p.l1w = nextp(); p.l1b = nextp();
      p.l2w = nextp(); p.l2b = nextp();
      p.l3w = nextp(); p.l3b = nextp();
    }
    for (int b = 0; b < 11; ++b) { bn[b].gamma = nextp(); bn[b].beta = nextp(); }
    segw = nextp(); segb = nextp(); clsw = nextp(); clsb = nextp();
  } else {
    for (int j = 0; j < 11; ++j) { int b = bnOrd[j]; bn[b].beta = nextp(); bn[b].gamma = nextp(); }
    clsb = nextp(); clsw = nextp();
    for (int j = 0; j < 11; ++j) {
      PCp& p = pc[cvOrd[j]];
      p.centers = nextp();
      p.l1b = nextp(); p.l1w = nextp();
      p.l2b = nextp(); p.l2w = nextp();
      p.l3b = nextp(); p.l3w = nextp();
      p.weight = nextp();
    }
    segb = nextp(); segw = nextp();
    points = nextp();
  }

  // ---- workspace carving ----
  char* wsb = (char*)d_ws;
  size_t off = 0;
  auto carve = [&](size_t bytes) -> void* {
    size_t o = (off + 255) & ~(size_t)255;
    off = o + bytes;
    return (void*)(wsb + o);
  };
  Scratch w;
  w.Pts  = (float*)carve((size_t)BB * NN * 3 * 4);
  w.idx  = (int*)carve((size_t)320000 * 4);
  w.D    = (half_t*)carve((size_t)320000 * 64 * 2);
  w.H1   = (half_t*)carve((size_t)320000 * 32 * 2);
  w.H2   = (half_t*)carve((size_t)320000 * 32 * 2);
  w.H    = (float*)carve((size_t)320000 * 16 * 4);
  w.F    = (half_t*)carve((size_t)20000 * 4864 * 2);
  w.RAW  = (float*)carve((size_t)20000 * 256 * 4);
  w.Bp1  = (half_t*)carve((size_t)4 * 512 * 2);
  w.Bp2  = (half_t*)carve((size_t)512 * 2);
  w.Bp3  = (half_t*)carve((size_t)512 * 2);
  w.BpW  = (half_t*)carve((size_t)1216 * 512 * 2);
  w.mean = (float*)carve((size_t)512 * 4);
  w.rstd = (float*)carve((size_t)512 * 4);
  float* X2   = (float*)carve((size_t)20000 * 48 * 4);
  float* X3   = (float*)carve((size_t)8192 * 48 * 4);
  float* X4   = (float*)carve((size_t)4096 * 96 * 4);
  float* X5   = (float*)carve((size_t)2048 * 96 * 4);
  float* X6   = (float*)carve((size_t)1024 * 128 * 4);
  float* X5D  = (float*)carve((size_t)2048 * 352 * 4);
  float* X4D  = (float*)carve((size_t)4096 * 192 * 4);
  float* X3D  = (float*)carve((size_t)8192 * 96 * 4);
  float* X2D  = (float*)carve((size_t)20000 * 304 * 4);
  float* SEGF = (float*)carve((size_t)20000 * 128 * 4);
  half_t* SEGF16 = (half_t*)carve((size_t)20000 * 128 * 2);
  float* SEGOUT  = (float*)carve((size_t)20000 * 64 * 4);
  float* CLSF    = (float*)carve((size_t)2 * 256 * 4);
  half_t* CLSF16 = (half_t*)carve((size_t)32 * 256 * 2);

  // ---- network ----
  k_transpose_pts<<<cdiv(BB * NN, 256), 256, 0, stream>>>(points, w.Pts);

  // encoder
  run_ptconv(stream, w, 10000, 10000, 16, (const float*)nullptr, 1, pc[0], 48, w.RAW);
  run_bn(stream, w, w.RAW, 20000, 48, bn[0], X2, 48, 0);

  run_ptconv(stream, w, 10000, 4096, 16, X2, 48, pc[1], 48, w.RAW);
  run_bn(stream, w, w.RAW, 8192, 48, bn[1], X3, 48, 0);

  run_ptconv(stream, w, 4096, 2048, 8, X3, 48, pc[2], 96, w.RAW);
  run_bn(stream, w, w.RAW, 4096, 96, bn[2], X4, 96, 0);

  run_ptconv(stream, w, 2048, 1024, 8, X4, 96, pc[3], 96, w.RAW);
  run_bn(stream, w, w.RAW, 2048, 96, bn[3], X5, 96, 0);

  run_ptconv(stream, w, 1024, 512, 4, X5, 96, pc[4], 128, w.RAW);
  run_bn(stream, w, w.RAW, 1024, 128, bn[4], X6, 128, 0);

  // decoder with skip concats
  run_ptconv(stream, w, 512, 1024, 4, X6, 128, pc[5], 256, w.RAW);
  run_bn(stream, w, w.RAW, 2048, 256, bn[5], X5D, 352, 0);
  k_copy_cols<<<cdiv(2048 * 96, 256), 256, 0, stream>>>(X5, 2048, 96, X5D, 352, 256);

  run_ptconv(stream, w, 1024, 2048, 4, X5D, 352, pc[6], 96, w.RAW);
  run_bn(stream, w, w.RAW, 4096, 96, bn[6], X4D, 192, 0);
  k_copy_cols<<<cdiv(4096 * 96, 256), 256, 0, stream>>>(X4, 4096, 96, X4D, 192, 96);

  run_ptconv(stream, w, 2048, 4096, 4, X4D, 192, pc[7], 48, w.RAW);
  run_bn(stream, w, w.RAW, 8192, 48, bn[7], X3D, 96, 0);
  k_copy_cols<<<cdiv(8192 * 48, 256), 256, 0, stream>>>(X3, 8192, 48, X3D, 96, 48);

  run_ptconv(stream, w, 4096, 10000, 8, X3D, 96, pc[8], 256, w.RAW);
  run_bn(stream, w, w.RAW, 20000, 256, bn[8], X2D, 304, 0);
  k_copy_cols<<<cdiv(20000 * 48, 256), 256, 0, stream>>>(X2, 20000, 48, X2D, 304, 256);

  run_ptconv(stream, w, 10000, 10000, 8, X2D, 304, pc[9], 128, w.RAW);
  run_bn(stream, w, w.RAW, 20000, 128, bn[9], SEGF, 128, 0);

  // segmentation head: [20000 x 128] x segw^T [128 x 50] + b -> transpose
  k_f32_to_f16<<<cdiv(20000 * 128, 256), 256, 0, stream>>>(SEGF, 20000, 128,
                                                           SEGF16, 20000, 128);
  k_pack_b<<<cdiv(4 * 4 * 512, 256), 256, 0, stream>>>(segw, 128, 50, 1, 4, 4, w.BpW);
  k_gemm_wmma<<<dim3(625, 1), 32, 0, stream>>>(SEGF16, 128, w.BpW, 4, 4,
      SEGOUT, 50, 20000, 50, (half_t*)nullptr, 0, segb, 1.0f, 0);
  k_seg_transpose<<<cdiv(BB * 50 * NN, 256), 256, 0, stream>>>(SEGOUT, (float*)d_out);

  // classification head: conv_cls on x6g with M=1, then linear to 16 classes
  run_ptconv(stream, w, 512, 1, 4, X6, 128, pc[10], 256, w.RAW);
  run_bn(stream, w, w.RAW, 2, 256, bn[10], CLSF, 256, 0);
  k_f32_to_f16<<<cdiv(32 * 256, 256), 256, 0, stream>>>(CLSF, 2, 256, CLSF16, 32, 256);
  k_pack_b<<<cdiv(8 * 512, 256), 256, 0, stream>>>(clsw, 256, 16, 1, 8, 1, w.BpW);
  k_gemm_wmma<<<dim3(1, 1), 32, 0, stream>>>(CLSF16, 256, w.BpW, 8, 1,
      (float*)d_out + (size_t)BB * 50 * NN, 16, 2, 16, (half_t*)nullptr, 0,
      clsb, 1.0f, 0);
}
